// MultiLayerRecurrenceInjector_22668837388382
// MI455X (gfx1250) — compile-verified, matrix operand order checked by reading, not run on hardware
//
#include <hip/hip_runtime.h>
#include <hip/hip_bf16.h>

// ---------------------------------------------------------------------------
// MI455X (gfx1250) implementation: bf16 WMMA GEMMs + flash attention,
// with CDNA5 async global->LDS copies for straight-copy tiles.
// ---------------------------------------------------------------------------

#define DEVI __device__ __forceinline__

typedef __attribute__((ext_vector_type(16))) __bf16       v16bf;
typedef __attribute__((ext_vector_type(8)))  float        v8f;
typedef __attribute__((ext_vector_type(4)))  float        vf4;
typedef __attribute__((ext_vector_type(4)))  unsigned int vu4;
typedef __attribute__((ext_vector_type(2)))  unsigned int vu2;
typedef unsigned short bfu;   // bf16 storage

constexpr int BB   = 4;
constexpr int SS   = 2048;
constexpr int HH   = 2048;
constexpr int KSEL = 1024;
constexpr int LL   = 2;
constexpr int NHH  = 16;
constexpr int FFF  = 4096;
constexpr int HDIM = HH / NHH;      // 128
constexpr int MROWS = BB * KSEL;    // 4096

union FragU { vu4 u[2]; v16bf v; };
union H8   { vu4 u; bfu h[8]; };
union H4   { vu2 u; bfu h[4]; };

DEVI bfu f2bf(float f) {                 // round-to-nearest-even fp32 -> bf16
  unsigned u = __float_as_uint(f);
  unsigned r = u + 0x7FFFu + ((u >> 16) & 1u);
  return (bfu)(r >> 16);
}

DEVI v8f vzero8() { v8f z = {0.f,0.f,0.f,0.f,0.f,0.f,0.f,0.f}; return z; }

// CDNA5 async copy: 16B global -> LDS, tracked by ASYNCcnt (no VGPR round-trip).
// GVS addressing: mem = SGPR base + 32-bit VGPR byte offset.
DEVI void async_ld_b128(void* ldsDst, const void* gbase, unsigned byteOff) {
  unsigned l = (unsigned)(uintptr_t)ldsDst;   // LDS window is low 32 bits
  asm volatile("global_load_async_to_lds_b128 %0, %1, %2"
               :: "v"(l), "v"(byteOff), "s"(gbase)
               : "memory");
}
DEVI void wait_async0() { asm volatile("s_wait_asynccnt 0x0" ::: "memory"); }

// Load one 16x32 bf16 A/B fragment (wave32 layout: lane<16 -> K {0..7,16..23},
// lane>=16 -> K {8..15,24..31}) from a row-major bf16 matrix (stride in elems).
DEVI v16bf load_frag(const bfu* base, int row, int k0, int stride) {
  int lane = threadIdx.x & 31;
  int r    = lane & 15;
  int kk   = k0 + ((lane & 16) ? 8 : 0);
  const bfu* p = base + (size_t)(row + r) * stride + kk;
  FragU f;
  f.u[0] = *(const vu4*)(p);
  f.u[1] = *(const vu4*)(p + 16);
  return f.v;
}

DEVI v8f wmma_bf16(v16bf a, v16bf b, v8f c) {
  return __builtin_amdgcn_wmma_f32_16x16x32_bf16(false, a, false, b,
                                                 (short)0, c, false, false);
}

// ---------------------------------------------------------------------------
// Gather: cur = x[idx] (f32), xin = bf16(cur), mem = bf16(prev[idx])
// ---------------------------------------------------------------------------
__global__ __launch_bounds__(256)
void gather_kernel(const float* __restrict__ x, const float* __restrict__ prev,
                   const int* __restrict__ idx, float* __restrict__ cur,
                   bfu* __restrict__ xin, bfu* __restrict__ mem) {
  int bk = blockIdx.x;                 // 0..MROWS-1
  int b = bk >> 10, kk = bk & (KSEL - 1);
  int srow = idx[b * KSEL + kk];
  size_t src = ((size_t)b * SS + srow) * HH;
  size_t dst = (size_t)bk * HH;
  int col = threadIdx.x * 8;
  vf4 x0 = *(const vf4*)(x + src + col);
  vf4 x1 = *(const vf4*)(x + src + col + 4);
  vf4 p0 = *(const vf4*)(prev + src + col);
  vf4 p1 = *(const vf4*)(prev + src + col + 4);
  *(vf4*)(cur + dst + col)     = x0;
  *(vf4*)(cur + dst + col + 4) = x1;
  H8 hx; hx.h[0]=f2bf(x0.x); hx.h[1]=f2bf(x0.y); hx.h[2]=f2bf(x0.z); hx.h[3]=f2bf(x0.w);
         hx.h[4]=f2bf(x1.x); hx.h[5]=f2bf(x1.y); hx.h[6]=f2bf(x1.z); hx.h[7]=f2bf(x1.w);
  *(vu4*)(xin + dst + col) = hx.u;
  H8 hp; hp.h[0]=f2bf(p0.x); hp.h[1]=f2bf(p0.y); hp.h[2]=f2bf(p0.z); hp.h[3]=f2bf(p0.w);
         hp.h[4]=f2bf(p1.x); hp.h[5]=f2bf(p1.y); hp.h[6]=f2bf(p1.z); hp.h[7]=f2bf(p1.w);
  *(vu4*)(mem + dst + col) = hp.u;
}

// ---------------------------------------------------------------------------
// GEMM: C[M,N] (+)= A[M,K](bf16) @ W[K,N](fp32, converted to bf16 in loader)
// MODE 0: C=f32 store, 1: C=f32 accumulate, 2: C=bf16 store
// Block tile 128x128x64, 256 threads (8 waves in 2x4), 4x2 frags per wave.
// A tile copied via async global->LDS; W tile converted through VGPRs.
// ---------------------------------------------------------------------------
template<int MODE>
__global__ __launch_bounds__(256)
void gemm_kernel(const bfu* __restrict__ A, const float* __restrict__ W,
                 float* __restrict__ Cf, bfu* __restrict__ Cb,
                 int Mdim, int Kdim, int Ndim) {
  __shared__ __align__(16) bfu As[128 * 64];   // [m][k]
  __shared__ __align__(16) bfu Bs[128 * 64];   // [n][k] (transposed W tile)
  int tid  = threadIdx.x;
  int lane = tid & 31, wave = tid >> 5;
  int wm = wave >> 2, wn = wave & 3;
  int mBase = blockIdx.y * 128, nBase = blockIdx.x * 128;

  v8f acc[4][2];
  #pragma unroll
  for (int i = 0; i < 4; i++)
    #pragma unroll
    for (int j = 0; j < 2; j++) acc[i][j] = vzero8();

  int nK = Kdim >> 6;
  for (int kb = 0; kb < nK; kb++) {
    // A tile: 128x64 bf16 = 1024 16B chunks, 4 async copies per thread
    #pragma unroll
    for (int i = 0; i < 4; i++) {
      int c = tid + i * 256;
      int row = c >> 3, col = (c & 7) * 8;
      unsigned goff = (unsigned)(((size_t)(mBase + row) * Kdim + kb * 64 + col)
                                 * sizeof(bfu));
      async_ld_b128(&As[row * 64 + col], A, goff);
    }
    // W tile: 64x128 fp32, convert -> bf16, store transposed [n][k]
    {
      int row = tid >> 2;                 // 0..63 (k within tile)
      int colBase = (tid & 3) * 32;       // n within tile
      const float* wp = W + (size_t)(kb * 64 + row) * Ndim + nBase + colBase;
      #pragma unroll
      for (int j = 0; j < 8; j++) {
        vf4 f = *(const vf4*)(wp + j * 4);
        Bs[(colBase + j * 4 + 0) * 64 + row] = f2bf(f.x);
        Bs[(colBase + j * 4 + 1) * 64 + row] = f2bf(f.y);
        Bs[(colBase + j * 4 + 2) * 64 + row] = f2bf(f.z);
        Bs[(colBase + j * 4 + 3) * 64 + row] = f2bf(f.w);
      }
      if (kb + 1 < nK) {
        __builtin_prefetch(W + (size_t)((kb + 1) * 64 + row) * Ndim + nBase + colBase, 0, 1);
      }
    }
    wait_async0();
    __syncthreads();
    #pragma unroll
    for (int ks = 0; ks < 2; ks++) {
      v16bf a4[4], b2[2];
      #pragma unroll
      for (int mt = 0; mt < 4; mt++) a4[mt] = load_frag(As, wm * 64 + mt * 16, ks * 32, 64);
      #pragma unroll
      for (int nt = 0; nt < 2; nt++) b2[nt] = load_frag(Bs, wn * 32 + nt * 16, ks * 32, 64);
      #pragma unroll
      for (int mt = 0; mt < 4; mt++)
        #pragma unroll
        for (int nt = 0; nt < 2; nt++)
          acc[mt][nt] = wmma_bf16(a4[mt], b2[nt], acc[mt][nt]);
    }
    __syncthreads();
  }
  // epilogue (C layout: VGPR r -> row r + (lane<16?0:8), col = lane&15)
  #pragma unroll
  for (int mt = 0; mt < 4; mt++) {
    #pragma unroll
    for (int nt = 0; nt < 2; nt++) {
      int row0 = mBase + wm * 64 + mt * 16 + ((lane & 16) ? 8 : 0);
      int col  = nBase + wn * 32 + nt * 16 + (lane & 15);
      #pragma unroll
      for (int r = 0; r < 8; r++) {
        size_t off = (size_t)(row0 + r) * Ndim + col;
        float vv = acc[mt][nt][r];
        if (MODE == 0)      Cf[off] = vv;
        else if (MODE == 1) Cf[off] += vv;
        else                Cb[off] = f2bf(vv);
      }
    }
  }
}

// ---------------------------------------------------------------------------
// Flash attention per (b, head, 128-query tile). Non-causal ALiBi bias.
// LDS: Qs 32KB + KPs 16KB (K tile, re-used as P tile) + Vt 16KB = 64KB.
// Q and K tiles are copied with async global->LDS.
// ---------------------------------------------------------------------------
__global__ __launch_bounds__(256)
void attn_kernel(const bfu* __restrict__ q, const bfu* __restrict__ kmat,
                 const bfu* __restrict__ v, bfu* __restrict__ o) {
  __shared__ __align__(16) bfu Qs[128 * HDIM];  // [qrow][d]
  __shared__ __align__(16) bfu KPs[64 * HDIM];  // K tile [key][d]; later P [row][key]
  __shared__ __align__(16) bfu Vt[HDIM * 64];   // transposed V tile [d][key]

  int tid = threadIdx.x, lane = tid & 31, wave = tid >> 5;
  int b = blockIdx.x / NHH, h = blockIdx.x % NHH;
  int qBase = blockIdx.y * 128;
  const bfu* qp = q    + (size_t)b * KSEL * HH + h * HDIM;
  const bfu* kp = kmat + (size_t)b * KSEL * HH + h * HDIM;
  const bfu* vp = v    + (size_t)b * KSEL * HH + h * HDIM;

  // load Q tile 128x128 (2048 16B chunks, 8 async copies per thread)
  #pragma unroll
  for (int i = 0; i < 8; i++) {
    int c = tid + i * 256;
    int row = c >> 4, col = (c & 15) * 8;
    unsigned goff = (unsigned)(((size_t)(qBase + row) * HH + col) * sizeof(bfu));
    async_ld_b128(&Qs[row * HDIM + col], qp, goff);
  }

  const float scale = 0.08838834764831845f;          // 1/sqrt(128)
  const float slope = exp2f(-0.5f * (float)(h + 1)); // ALiBi slope

  float mI[8], lI[8];
  v8f Ofr[8];
  #pragma unroll
  for (int r = 0; r < 8; r++) { mI[r] = -1e30f; lI[r] = 0.f; }
  #pragma unroll
  for (int d = 0; d < 8; d++) Ofr[d] = vzero8();

  int rowHalf = (lane & 16) ? 8 : 0;

  for (int kt = 0; kt < KSEL / 64; kt++) {
    int ktBase = kt * 64;
    __syncthreads();  // previous tile's P/V reads complete
    // load K tile 64x128 (1024 chunks, 4 async copies per thread)
    #pragma unroll
    for (int i = 0; i < 4; i++) {
      int c = tid + i * 256;
      int row = c >> 4, col = (c & 15) * 8;
      unsigned goff = (unsigned)(((size_t)(ktBase + row) * HH + col) * sizeof(bfu));
      async_ld_b128(&KPs[row * HDIM + col], kp, goff);
    }
    // load V tile transposed: Vt[d][key] (needs VGPRs for the transpose)
    {
      int key = tid >> 2;
      int dBase = (tid & 3) * 32;
      #pragma unroll
      for (int j = 0; j < 4; j++) {
        H8 hv; hv.u = *(const vu4*)(vp + (size_t)(ktBase + key) * HH + dBase + j * 8);
        #pragma unroll
        for (int e = 0; e < 8; e++) Vt[(dBase + j * 8 + e) * 64 + key] = hv.h[e];
      }
    }
    wait_async0();    // Q tile (first iter) + K tile landed in LDS
    __syncthreads();

    // S = Q * K^T : 4 n-frags (16 keys each), 4 k-steps over d
    v16bf a4[4];
    #pragma unroll
    for (int ks = 0; ks < 4; ks++) a4[ks] = load_frag(Qs, wave * 16, ks * 32, HDIM);
    v8f s[4];
    #pragma unroll
    for (int nt = 0; nt < 4; nt++) {
      v8f c = vzero8();
      #pragma unroll
      for (int ks = 0; ks < 4; ks++) {
        v16bf bfr = load_frag(KPs, nt * 16, ks * 32, HDIM);
        c = wmma_bf16(a4[ks], bfr, c);
      }
      s[nt] = c;
    }
    // scale + ALiBi bias
    #pragma unroll
    for (int nt = 0; nt < 4; nt++) {
      int kcol = ktBase + nt * 16 + (lane & 15);
      #pragma unroll
      for (int r = 0; r < 8; r++) {
        int qrow = qBase + wave * 16 + rowHalf + r;
        s[nt][r] = s[nt][r] * scale - slope * fabsf((float)(qrow - kcol));
      }
    }
    // online softmax (row reductions over the 16-lane half via shfl_xor)
    float corr[8];
    #pragma unroll
    for (int r = 0; r < 8; r++) {
      float mx = s[0][r];
      #pragma unroll
      for (int nt = 1; nt < 4; nt++) mx = fmaxf(mx, s[nt][r]);
      #pragma unroll
      for (int msk = 8; msk >= 1; msk >>= 1) mx = fmaxf(mx, __shfl_xor(mx, msk, 32));
      float mNew = fmaxf(mI[r], mx);
      corr[r] = __expf(mI[r] - mNew);
      mI[r] = mNew;
      float rs = 0.f;
      #pragma unroll
      for (int nt = 0; nt < 4; nt++) {
        float p = __expf(s[nt][r] - mNew);
        s[nt][r] = p;
        rs += p;
      }
      #pragma unroll
      for (int msk = 8; msk >= 1; msk >>= 1) rs += __shfl_xor(rs, msk, 32);
      lI[r] = lI[r] * corr[r] + rs;
    }
    #pragma unroll
    for (int d = 0; d < 8; d++)
      #pragma unroll
      for (int r = 0; r < 8; r++) Ofr[d][r] = Ofr[d][r] * corr[r];

    __syncthreads();  // all waves done reading K tile -> reuse as P tile
    // write P (bf16, A layout source) into KPs as Ps[row][key]
    #pragma unroll
    for (int nt = 0; nt < 4; nt++) {
      int kcolL = nt * 16 + (lane & 15);
      #pragma unroll
      for (int r = 0; r < 8; r++)
        KPs[(wave * 16 + rowHalf + r) * 64 + kcolL] = f2bf(s[nt][r]);
    }
    // O += P * V  (K-dim = 64 keys -> 2 k-steps; per-wave LDS RAW is in-order)
    #pragma unroll
    for (int ks2 = 0; ks2 < 2; ks2++) {
      v16bf pa = load_frag(KPs, wave * 16, ks2 * 32, 64);
      #pragma unroll
      for (int d = 0; d < 8; d++) {
        v16bf vfr = load_frag(Vt, d * 16, ks2 * 32, 64);
        Ofr[d] = wmma_bf16(pa, vfr, Ofr[d]);
      }
    }
  }
  // finalize: O /= l, store bf16
  #pragma unroll
  for (int r = 0; r < 8; r++) {
    float inv = 1.f / lI[r];
    int grow = qBase + wave * 16 + rowHalf + r;
    bfu* op = o + (size_t)(b * KSEL + grow) * HH + h * HDIM + (lane & 15);
    #pragma unroll
    for (int d = 0; d < 8; d++) op[d * 16] = f2bf(Ofr[d][r] * inv);
  }
}

// ---------------------------------------------------------------------------
// LayerNorm. MODE 1: bf16 out only. MODE 2: f32 in-place + bf16 out.
// ---------------------------------------------------------------------------
template<int MODE>
__global__ __launch_bounds__(256)
void ln_kernel(const float* __restrict__ in, const float* __restrict__ gamma,
               const float* __restrict__ beta, float* __restrict__ outf,
               bfu* __restrict__ outb) {
  __shared__ float red[16];
  int row = blockIdx.x, tid = threadIdx.x;
  int lane = tid & 31, wave = tid >> 5;
  const float* p = in + (size_t)row * HH + tid * 8;
  vf4 v0 = *(const vf4*)p;
  vf4 v1 = *(const vf4*)(p + 4);
  float a[8] = {v0.x, v0.y, v0.z, v0.w, v1.x, v1.y, v1.z, v1.w};
  float s = 0.f, s2 = 0.f;
  #pragma unroll
  for (int j = 0; j < 8; j++) { s += a[j]; s2 += a[j] * a[j]; }
  #pragma unroll
  for (int m = 16; m >= 1; m >>= 1) {
    s  += __shfl_xor(s,  m, 32);
    s2 += __shfl_xor(s2, m, 32);
  }
  if (lane == 0) { red[wave] = s; red[8 + wave] = s2; }
  __syncthreads();
  if (tid == 0) {
    float ts = 0.f, ts2 = 0.f;
    for (int w2 = 0; w2 < 8; w2++) { ts += red[w2]; ts2 += red[8 + w2]; }
    red[0] = ts; red[8] = ts2;
  }
  __syncthreads();
  float mean = red[0] * (1.f / (float)HH);
  float var  = red[8] * (1.f / (float)HH) - mean * mean;
  float rstd = rsqrtf(var + 1e-5f);
  int col = tid * 8;
  float o[8]; H8 ob;
  #pragma unroll
  for (int j = 0; j < 8; j++) {
    float nv = (a[j] - mean) * rstd * gamma[col + j] + beta[col + j];
    o[j] = nv; ob.h[j] = f2bf(nv);
  }
  if (MODE == 2) {
    vf4 w0 = {o[0], o[1], o[2], o[3]};
    vf4 w1 = {o[4], o[5], o[6], o[7]};
    *(vf4*)(outf + (size_t)row * HH + col)     = w0;
    *(vf4*)(outf + (size_t)row * HH + col + 4) = w1;
  }
  *(vu4*)(outb + (size_t)row * HH + col) = ob.u;
}

// ---------------------------------------------------------------------------
// ffh = bf16(silu(g) * u)
// ---------------------------------------------------------------------------
__global__ __launch_bounds__(256)
void silu_mul_kernel(const float* __restrict__ g, const float* __restrict__ u,
                     bfu* __restrict__ h) {
  size_t i = ((size_t)blockIdx.x * 256 + threadIdx.x) * 4;
  vf4 gv = *(const vf4*)(g + i);
  vf4 uv = *(const vf4*)(u + i);
  H4 o;
  o.h[0] = f2bf(gv.x / (1.f + __expf(-gv.x)) * uv.x);
  o.h[1] = f2bf(gv.y / (1.f + __expf(-gv.y)) * uv.y);
  o.h[2] = f2bf(gv.z / (1.f + __expf(-gv.z)) * uv.z);
  o.h[3] = f2bf(gv.w / (1.f + __expf(-gv.w)) * uv.w);
  *(vu2*)(h + i) = o.u;
}

__global__ __launch_bounds__(256)
void copy_kernel(const float* __restrict__ x, float* __restrict__ out) {
  size_t i = ((size_t)blockIdx.x * 256 + threadIdx.x) * 4;
  *(vf4*)(out + i) = *(const vf4*)(x + i);
}

__global__ __launch_bounds__(256)
void scatter_kernel(const float* __restrict__ cur, const int* __restrict__ idx,
                    float* __restrict__ out) {
  int bk = blockIdx.x;
  int b = bk >> 10, kk = bk & (KSEL - 1);
  int drow = idx[b * KSEL + kk];
  size_t dst = ((size_t)b * SS + drow) * HH;
  size_t src = (size_t)bk * HH;
  int col = threadIdx.x * 8;
  vf4 t0 = *(vf4*)(out + dst + col);
  vf4 t1 = *(vf4*)(out + dst + col + 4);
  t0 += *(const vf4*)(cur + src + col);
  t1 += *(const vf4*)(cur + src + col + 4);
  *(vf4*)(out + dst + col)     = t0;
  *(vf4*)(out + dst + col + 4) = t1;
}

// ---------------------------------------------------------------------------
extern "C" void kernel_launch(void* const* d_in, const int* in_sizes, int n_in,
                              void* d_out, int out_size, void* d_ws, size_t ws_size,
                              hipStream_t stream) {
  (void)in_sizes; (void)n_in; (void)out_size; (void)ws_size;
  const float* x     = (const float*)d_in[0];
  const float* prev  = (const float*)d_in[1];
  const int*   idx   = (const int*)d_in[2];
  const float* Wq    = (const float*)d_in[3];
  const float* Wk    = (const float*)d_in[4];
  const float* Wv    = (const float*)d_in[5];
  const float* Wo    = (const float*)d_in[6];
  const float* ag    = (const float*)d_in[7];
  const float* abta  = (const float*)d_in[8];
  const float* mg    = (const float*)d_in[9];
  const float* mbta  = (const float*)d_in[10];
  const float* Wgate = (const float*)d_in[11];
  const float* Wup   = (const float*)d_in[12];
  const float* Wdown = (const float*)d_in[13];
  float* out = (float*)d_out;
  char* ws = (char*)d_ws;

  constexpr size_t MB = 1024ull * 1024ull;
  float* cur = (float*)(ws + 0);          //  32 MB  fp32 residual stream
  bfu*   mem = (bfu*)(ws + 32 * MB);      //  16 MB  bf16(prev[idx])
  bfu*   xin = (bfu*)(ws + 48 * MB);      //  16 MB  bf16 GEMM input / normed
  bfu*   qb  = (bfu*)(ws + 64 * MB);      //  16 MB
  bfu*   kb  = (bfu*)(ws + 80 * MB);      //  16 MB
  bfu*   vb  = (bfu*)(ws + 96 * MB);      //  16 MB
  bfu*   ao  = (bfu*)(ws + 112 * MB);     //  16 MB  attention output
  float* ff1 = (float*)(ws + 128 * MB);   //  64 MB  gate
  float* ff2 = (float*)(ws + 192 * MB);   //  64 MB  up
  bfu*   ffh = (bfu*)(ws + 256 * MB);     //  32 MB  silu(g)*u  (total 288 MB)

  gather_kernel<<<MROWS, 256, 0, stream>>>(x, prev, idx, cur, xin, mem);

  dim3 gH(HH / 128, MROWS / 128);    // N=2048 GEMMs
  dim3 gF(FFF / 128, MROWS / 128);   // N=4096 GEMMs
  size_t HH2 = (size_t)HH * HH;
  size_t HF  = (size_t)HH * FFF;

  for (int l = 0; l < LL; l++) {
    const float* wq = Wq + l * HH2;
    const float* wk = Wk + l * HH2;
    const float* wv = Wv + l * HH2;
    const float* wo = Wo + l * HH2;
    const float* wg = Wgate + l * HF;
    const float* wu = Wup + l * HF;
    const float* wd = Wdown + l * HF;

    gemm_kernel<2><<<gH, 256, 0, stream>>>(xin, wq, nullptr, qb, MROWS, HH, HH);
    gemm_kernel<2><<<gH, 256, 0, stream>>>(mem, wk, nullptr, kb, MROWS, HH, HH);
    gemm_kernel<2><<<gH, 256, 0, stream>>>(mem, wv, nullptr, vb, MROWS, HH, HH);
    attn_kernel<<<dim3(BB * NHH, KSEL / 128), 256, 0, stream>>>(qb, kb, vb, ao);
    gemm_kernel<1><<<gH, 256, 0, stream>>>(ao, wo, cur, nullptr, MROWS, HH, HH);
    ln_kernel<1><<<MROWS, 256, 0, stream>>>(cur, ag + l * HH, abta + l * HH, nullptr, xin);
    gemm_kernel<0><<<gF, 256, 0, stream>>>(xin, wg, ff1, nullptr, MROWS, HH, FFF);
    gemm_kernel<0><<<gF, 256, 0, stream>>>(xin, wu, ff2, nullptr, MROWS, HH, FFF);
    silu_mul_kernel<<<(MROWS * FFF) / 1024, 256, 0, stream>>>(ff1, ff2, ffh);
    gemm_kernel<1><<<gH, 256, 0, stream>>>(ffh, wd, cur, nullptr, MROWS, FFF, HH);
    ln_kernel<2><<<MROWS, 256, 0, stream>>>(cur, mg + l * HH, mbta + l * HH, cur, xin);
  }

  copy_kernel<<<(BB * SS * HH) / 1024, 256, 0, stream>>>(x, out);
  scatter_kernel<<<MROWS, 256, 0, stream>>>(cur, idx, out);
}
